// Net_1004_1288490189579
// MI455X (gfx1250) — matrix operand order repaired and checked by reading, not compiled, not run on hardware
//
#include <hip/hip_runtime.h>
#include <cstdint>
#include <cstddef>

#define NB 32
#define NNODE 400
#define NTOT (NB * NNODE)   // 12800
#define DINX 64
#define HH 128
#define DOUTX 64
#define TT 6
#define HORZ 3
#define EE 160000

typedef __attribute__((ext_vector_type(16))) _Float16 v16h;
typedef __attribute__((ext_vector_type(8)))  _Float16 v8h;
typedef __attribute__((ext_vector_type(8)))  float    v8f;

// ---------------- elementwise / prep kernels ----------------

__global__ void fill_zero_f32(float* __restrict__ p, int n) {
  int t = blockIdx.x * blockDim.x + threadIdx.x;
  if (t < n) p[t] = 0.0f;
}

// Pack a (K x Hout) row-major f32 weight into WMMA B-fragment order (f16):
// fragment (kt,ct): 32 lanes x 16 halves, lane L -> col ct*16+(L&15),
// half = L>>4: elements 0-7 -> K kt*32 + half*8 + e, elements 8-15 -> K kt*32 + 16 + half*8 + (e-8).
__global__ void pack_weight_f16(const float* __restrict__ W, _Float16* __restrict__ out,
                                int K, int Hout) {
  int tid = blockIdx.x * blockDim.x + threadIdx.x;
  int total = K * Hout;
  if (tid >= total) return;
  int e  = tid & 15;
  int L  = (tid >> 4) & 31;
  int fi = tid >> 9;                 // fragment index = kt*ntiles + ct
  int ntiles = Hout >> 4;
  int kt = fi / ntiles;
  int ct = fi - kt * ntiles;
  int col  = ct * 16 + (L & 15);
  int half = L >> 4;
  int k = kt * 32 + (e < 8 ? half * 8 + e : 16 + half * 8 + (e - 8));
  out[tid] = (_Float16)W[(size_t)k * Hout + col];
}

// zh = concat( prelu?(x), rbuf ? rbuf*h : h ), and zero the agg buffer.
__global__ void build_concat(const float* __restrict__ x, int dx,
                             const float* __restrict__ h,
                             const float* __restrict__ rbuf,
                             const float* __restrict__ alpha,
                             float* __restrict__ zh, float* __restrict__ agg,
                             int din, int n) {
  int tid = blockIdx.x * blockDim.x + threadIdx.x;
  if (tid >= n * din) return;
  int i = tid / din;
  int j = tid - i * din;
  float v;
  if (j < dx) {
    v = x[(size_t)i * dx + j];
    if (alpha) v = (v >= 0.0f) ? v : alpha[j] * v;
  } else {
    int jj = j - dx;
    float hv = h[(size_t)i * HH + jj];
    v = rbuf ? rbuf[(size_t)i * HH + jj] * hv : hv;
  }
  zh[tid] = v;
  agg[tid] = 0.0f;
}

// agg[dst[e]] += zh[src[e]] * ea[e]   (segment_sum via hw fp32 atomics; L2-resident tables)
__global__ void edge_aggregate(const float* __restrict__ zh,
                               const int* __restrict__ src,
                               const int* __restrict__ dst,
                               const float* __restrict__ ea,
                               float* __restrict__ agg, int din) {
  int cpe = din >> 2;                        // float4 chunks per edge
  int tid = blockIdx.x * blockDim.x + threadIdx.x;
  if (tid >= EE * cpe) return;
  int e = tid / cpe;
  int c = tid - e * cpe;
  int s = src[e], d = dst[e];
  float w = ea[e];
  float4 v = reinterpret_cast<const float4*>(zh + (size_t)s * din)[c];
  float* o = agg + (size_t)d * din + (size_t)c * 4;
  unsafeAtomicAdd(o + 0, v.x * w);
  unsafeAtomicAdd(o + 1, v.y * w);
  unsafeAtomicAdd(o + 2, v.z * w);
  unsafeAtomicAdd(o + 3, v.w * w);
}

__global__ void add_cvt_f16(const float* __restrict__ a, const float* __restrict__ b,
                            _Float16* __restrict__ o, int n) {
  int t = blockIdx.x * blockDim.x + threadIdx.x;
  if (t < n) o[t] = (_Float16)(a[t] + b[t]);
}

__global__ void cvt_f16(const float* __restrict__ a, _Float16* __restrict__ o, int n) {
  int t = blockIdx.x * blockDim.x + threadIdx.x;
  if (t < n) o[t] = (_Float16)a[t];
}

// ---------------- WMMA GEMM with fused epilogues ----------------
// One wave per (16-row tile) x (CPW 16-col tiles); 4 waves/block, grid.x == 1.
// A fragment loaded once per K-step and reused for CPW WMMAs (register blocking).
// mode 0: out0 = sigmoid(A@B + bias)
// mode 1: hBuf = u*hBuf + (1-u)*tanh(A@B + bias)     (GRU update, in place)
// mode 2: pred = A@B + bias -> xinBuf (feedback) and gOut at (B,HOR,NN,DOUT) transpose
template <int CPW>
__global__ __launch_bounds__(128)
void gemm_wmma(const _Float16* __restrict__ A, const _Float16* __restrict__ Bp,
               const float* __restrict__ bias, int K, int Hout, int mode,
               float* __restrict__ out0, const float* __restrict__ uBuf,
               float* __restrict__ hBuf, float* __restrict__ xinBuf,
               float* __restrict__ gOut, int step)
{
  const int lane = threadIdx.x & 31;
  const int wv   = threadIdx.x >> 5;
  const int ntiles = Hout >> 4;
  const int ct0 = wv * CPW;
  if (ct0 >= ntiles) return;                 // wave-uniform: EXEC stays all-1s
  const int rt   = blockIdx.y;
  const int half = lane >> 4;
  const int row  = rt * 16 + (lane & 15);
  const int ktiles = K >> 5;

  const _Float16* arow = A + (size_t)row * K;
  const _Float16* bbase = Bp + ((size_t)lane) * 16;
  const size_t bstride = (size_t)ntiles * 512;   // halves per kt step

  v8f acc[CPW];
#pragma unroll
  for (int j = 0; j < CPW; ++j) acc[j] = (v8f){};

  for (int kt = 0; kt < ktiles; ++kt) {
    // A fragment: lane holds row (lane&15); K slices 0-7/16-23 (half 0) or 8-15/24-31 (half 1)
    union AU { v16h v; v8h h2[2]; } a;
    const _Float16* ap = arow + kt * 32 + half * 8;
    a.h2[0] = *reinterpret_cast<const v8h*>(ap);
    a.h2[1] = *reinterpret_cast<const v8h*>(ap + 16);
#pragma unroll
    for (int j = 0; j < CPW; ++j) {
      // B fragment: pre-packed, 32 contiguous bytes per lane
      v16h b = *reinterpret_cast<const v16h*>(
          bbase + (size_t)kt * bstride + (size_t)(ct0 + j) * 512);
      acc[j] = __builtin_amdgcn_wmma_f32_16x16x32_f16(false, a.v, false, b,
                                                      (short)0, acc[j], false, false);
    }
  }

  const int rbase = rt * 16 + half * 8;      // D: M = vgpr + 8*half, N = lane&15
#pragma unroll
  for (int j = 0; j < CPW; ++j) {
    const int col = (ct0 + j) * 16 + (lane & 15);
    const float bv = bias[col];
#pragma unroll
    for (int v = 0; v < 8; ++v) {
      const int r = rbase + v;
      const float val = acc[j][v] + bv;
      const size_t idx = (size_t)r * Hout + col;
      if (mode == 0) {
        out0[idx] = 1.0f / (1.0f + __expf(-val));
      } else if (mode == 1) {
        const float uu = uBuf[idx];
        hBuf[idx] = uu * hBuf[idx] + (1.0f - uu) * tanhf(val);
      } else {
        xinBuf[idx] = val;                   // feedback for next decoder step
        const int bb = r / NNODE;
        const int nn = r - bb * NNODE;
        gOut[(((size_t)bb * HORZ + step) * NNODE + nn) * DOUTX + col] = val;
      }
    }
  }
}

// ---------------- host orchestration ----------------

extern "C" void kernel_launch(void* const* d_in, const int* in_sizes, int n_in,
                              void* d_out, int out_size, void* d_ws, size_t ws_size,
                              hipStream_t stream)
{
  (void)in_sizes; (void)n_in; (void)out_size; (void)ws_size;

  const float* x_seq = (const float*)d_in[0];
  const int*   srcI  = (const int*)d_in[1];
  const int*   dstI  = (const int*)d_in[2];
  const float* ea    = (const float*)d_in[3];
  const float* alpha = (const float*)d_in[4];
  const float* WoutF = (const float*)d_in[5];
  const float* boutF = (const float*)d_in[6];

  const float* Wm[4][3];
  const float* bm[4][3];
  const int din[4] = { DINX + HH, HH + HH, DOUTX + HH, HH + HH };  // enc1,enc2,dec1,dec2
  for (int l = 0; l < 4; ++l) {
    int base = 7 + l * 6;                       // dict order: Wr,Wu,Wc,br,bu,bc
    for (int g = 0; g < 3; ++g) {
      Wm[l][g] = (const float*)d_in[base + g];
      bm[l][g] = (const float*)d_in[base + 3 + g];
    }
  }

  char* pws = (char*)d_ws;
  auto bump = [&](size_t bytes) -> void* {
    void* r = (void*)pws;
    pws += (bytes + 255) & ~(size_t)255;
    return r;
  };

  _Float16* pw[4][3];
  for (int l = 0; l < 4; ++l)
    for (int g = 0; g < 3; ++g)
      pw[l][g] = (_Float16*)bump((size_t)din[l] * HH * sizeof(_Float16));
  _Float16* pwout = (_Float16*)bump((size_t)HH * DOUTX * sizeof(_Float16));

  float*    zh  = (float*)bump((size_t)NTOT * 256 * sizeof(float));
  float*    agg = (float*)bump((size_t)NTOT * 256 * sizeof(float));
  _Float16* s16 = (_Float16*)bump((size_t)NTOT * 256 * sizeof(_Float16));
  float*    rb  = (float*)bump((size_t)NTOT * HH * sizeof(float));
  float*    ub  = (float*)bump((size_t)NTOT * HH * sizeof(float));
  float*    h1  = (float*)bump((size_t)NTOT * HH * sizeof(float));
  float*    h2  = (float*)bump((size_t)NTOT * HH * sizeof(float));
  _Float16* h16 = (_Float16*)bump((size_t)NTOT * HH * sizeof(_Float16));
  float*    xin = (float*)bump((size_t)NTOT * DOUTX * sizeof(float));

  // one-time weight packing (per launch; cheap)
  for (int l = 0; l < 4; ++l)
    for (int g = 0; g < 3; ++g) {
      int tot = din[l] * HH;
      pack_weight_f16<<<(tot + 255) / 256, 256, 0, stream>>>(Wm[l][g], pw[l][g], din[l], HH);
    }
  pack_weight_f16<<<(HH * DOUTX + 255) / 256, 256, 0, stream>>>(WoutF, pwout, HH, DOUTX);

  fill_zero_f32<<<(NTOT * HH + 255) / 256, 256, 0, stream>>>(h1, NTOT * HH);
  fill_zero_f32<<<(NTOT * HH + 255) / 256, 256, 0, stream>>>(h2, NTOT * HH);

  const dim3 gGemm(1, NTOT / 16);     // 4 waves/block x CPW col tiles covers Hout exactly

  auto cell = [&](const float* x, int dx, float* h, bool usePrelu, int l) {
    const int dn   = dx + HH;
    const int tot  = NTOT * dn;
    const int etot = EE * (dn >> 2);
    const float* al = usePrelu ? alpha : nullptr;

    // r, u gates
    build_concat<<<(tot + 255) / 256, 256, 0, stream>>>(x, dx, h, nullptr, al, zh, agg, dn, NTOT);
    edge_aggregate<<<(etot + 255) / 256, 256, 0, stream>>>(zh, srcI, dstI, ea, agg, dn);
    add_cvt_f16<<<(tot + 255) / 256, 256, 0, stream>>>(zh, agg, s16, tot);
    gemm_wmma<2><<<gGemm, 128, 0, stream>>>(s16, pw[l][0], bm[l][0], dn, HH, 0,
                                            rb, nullptr, nullptr, nullptr, nullptr, 0);
    gemm_wmma<2><<<gGemm, 128, 0, stream>>>(s16, pw[l][1], bm[l][1], dn, HH, 0,
                                            ub, nullptr, nullptr, nullptr, nullptr, 0);
    // candidate + GRU update fused into epilogue
    build_concat<<<(tot + 255) / 256, 256, 0, stream>>>(x, dx, h, rb, al, zh, agg, dn, NTOT);
    edge_aggregate<<<(etot + 255) / 256, 256, 0, stream>>>(zh, srcI, dstI, ea, agg, dn);
    add_cvt_f16<<<(tot + 255) / 256, 256, 0, stream>>>(zh, agg, s16, tot);
    gemm_wmma<2><<<gGemm, 128, 0, stream>>>(s16, pw[l][2], bm[l][2], dn, HH, 1,
                                            nullptr, ub, h, nullptr, nullptr, 0);
  };

  // encoder
  for (int t = 0; t < TT; ++t) {
    const float* xt = x_seq + (size_t)t * NTOT * DINX;
    cell(xt, DINX, h1, false, 0);
    cell(h1, HH, h2, true, 1);
  }

  // decoder (autoregressive)
  fill_zero_f32<<<(NTOT * DOUTX + 255) / 256, 256, 0, stream>>>(xin, NTOT * DOUTX);
  for (int t = 0; t < HORZ; ++t) {
    cell(xin, DOUTX, h1, false, 2);
    cell(h1, HH, h2, true, 3);
    cvt_f16<<<(NTOT * HH + 255) / 256, 256, 0, stream>>>(h2, h16, NTOT * HH);
    gemm_wmma<1><<<gGemm, 128, 0, stream>>>(h16, pwout, boutF, HH, DOUTX, 2,
                                            nullptr, nullptr, nullptr, xin, (float*)d_out, t);
  }
}
